// LFNet_28965259444725
// MI455X (gfx1250) — matrix operand order repaired and checked by reading, not compiled, not run on hardware
//
#include <hip/hip_runtime.h>
#include <hip/hip_bf16.h>
#include <math.h>

// ---------------------------------------------------------------------------
// LFNet-style detector+descriptor for gfx1250 (MI455X, wave32, WMMA).
// All GEMM-shaped work (residual 5x5 convs, desc convs via im2col, FCs) runs
// on v_wmma_f32_16x16x32_f16 with f32 accumulation; activations stored f16.
// Interior conv tiles stage their LDS halo via GLOBAL_LOAD_ASYNC_TO_LDS_B128
// (ASYNCcnt) -- the CDNA5 async memory->LDS path -- with s_wait_asynccnt.
// Scalar kernels cover resize / softmaxpool / NMS / deterministic top-k /
// patch crop. Compile-only environment: layouts follow CDNA5 ISA.
// ---------------------------------------------------------------------------

typedef _Float16 half_t;
typedef __attribute__((ext_vector_type(16))) _Float16 v16h;
typedef __attribute__((ext_vector_type(8)))  float    v8f;

#define DEV __device__ __forceinline__

static constexpr int B_   = 4;
static constexpr int H_   = 480;
static constexpr int W_   = 640;
static constexpr int HW_  = H_ * W_;
static constexpr int NS   = 5;
static constexpr int TOPK = 500;
static constexpr int NKP  = B_ * TOPK;   // 2000 keypoints total

__constant__ float c_sf[NS] = {1.41421356f, 1.18920712f, 1.0f, 0.84089642f, 0.70710678f};

DEV float leakyf(float x) { return x >= 0.f ? x : 0.2f * x; }
DEV float bnf(float x, float g, float b, float m, float v) {
  return (x - m) * rsqrtf(v + 1e-5f) * g + b;
}
DEV v8f wmma_f16(v16h a, v16h b, v8f c) {
  return __builtin_amdgcn_wmma_f32_16x16x32_f16(false, a, false, b, (short)0, c, false, false);
}

// ------------------------------- elementwise --------------------------------

__global__ void k_mean3(const float* __restrict__ x, float* __restrict__ data) {
  int i = blockIdx.x * blockDim.x + threadIdx.x;
  if (i >= B_ * HW_) return;
  int b = i / HW_, p = i % HW_;
  const float* xb = x + (size_t)b * 3 * HW_ + p;
  data[i] = (xb[0] + xb[HW_] + xb[2 * HW_]) * (1.f / 3.f);
}

// init conv: 1->16, 5x5, pad2, scalar (only ~1 GFLOP)
__global__ void k_conv_init(const float* __restrict__ data, const float* __restrict__ w,
                            const float* __restrict__ bias, float* __restrict__ f) {
  int i = blockIdx.x * blockDim.x + threadIdx.x;
  if (i >= B_ * HW_) return;
  int b = i / HW_, p = i % HW_, y = p / W_, x = p % W_;
  float acc[16];
  for (int o = 0; o < 16; ++o) acc[o] = bias[o];
  for (int ky = 0; ky < 5; ++ky) {
    int yy = y + ky - 2; if (yy < 0 || yy >= H_) continue;
    for (int kx = 0; kx < 5; ++kx) {
      int xx = x + kx - 2; if (xx < 0 || xx >= W_) continue;
      float v = data[(size_t)b * HW_ + yy * W_ + xx];
      for (int o = 0; o < 16; ++o) acc[o] += w[o * 25 + ky * 5 + kx] * v;
    }
  }
  for (int o = 0; o < 16; ++o) f[((size_t)(b * 16 + o)) * HW_ + p] = acc[o];
}

// bn + leaky, NCHW f32 -> NHWC f16 (feeds the WMMA conv)
__global__ void k_bn_act_nhwc(const float* __restrict__ src,
                              const float* g, const float* bb, const float* m, const float* v,
                              half_t* __restrict__ dst) {
  int i = blockIdx.x * blockDim.x + threadIdx.x;
  if (i >= B_ * 16 * HW_) return;
  int c = i & 15; int bp = i >> 4; int b = bp / HW_; int p = bp % HW_;
  float x = src[((size_t)(b * 16 + c)) * HW_ + p];
  dst[i] = (half_t)leakyf(bnf(x, g[c], bb[c], m[c], v[c]));
}

__global__ void k_add(float* __restrict__ f, const float* __restrict__ c, long n) {
  long i = (long)blockIdx.x * blockDim.x + threadIdx.x;
  if (i < n) f[i] += c[i];
}

__global__ void k_tof16(const float* __restrict__ s, half_t* __restrict__ d, long n) {
  long i = (long)blockIdx.x * blockDim.x + threadIdx.x;
  if (i < n) d[i] = (half_t)s[i];
}

// pack conv weights [16o][16i][5][5] -> [26 taps][16 n][16 c] f16 (tap 25 = 0)
__global__ void k_pack_w5(const float* __restrict__ w, half_t* __restrict__ wp) {
  int i = blockIdx.x * blockDim.x + threadIdx.x;
  if (i >= 26 * 256) return;
  int t = i >> 8, n = (i >> 4) & 15, c = i & 15;
  wp[i] = (half_t)((t < 25) ? w[(n * 16 + c) * 25 + t] : 0.f);
}

// --------------------- 16->16 5x5 conv as WMMA implicit GEMM ----------------
// per-wave: 4x4 output pixels x 16 outch; 8x8x16ch halo tile staged in LDS.
// Interior tiles use async global->LDS (ASYNCcnt); border tiles use the
// predicated VGPR path with zero fill.
__global__ void __launch_bounds__(256) k_conv5x5_wmma(
    const half_t* __restrict__ in,   // NHWC [B][H][W][16]
    const half_t* __restrict__ wp,   // [26][16 n][16 c]
    const float*  __restrict__ bias, // [16]
    float* __restrict__ out)         // NCHW [B][16][H][W]
{
  __shared__ __align__(16) half_t lds[8][8 * 8 * 16];
  const int wave = threadIdx.x >> 5;
  const int lane = threadIdx.x & 31;
  const int tilesX = W_ / 4, tilesY = H_ / 4;
  long wid = (long)blockIdx.x * 8 + wave;
  if (wid >= (long)B_ * tilesY * tilesX) return;
  int b  = (int)(wid / (tilesY * tilesX));
  int t  = (int)(wid % (tilesY * tilesX));
  int y0 = (t / tilesX) * 4, x0 = (t % tilesX) * 4;
  half_t* L = lds[wave];

  // wave-uniform interior test (all 64 halo positions in-image)
  bool interior = (y0 >= 2) && (y0 + 5 < H_) && (x0 >= 2) && (x0 + 5 < W_);
  if (interior) {
    // async global->LDS: 64 positions (8x8) x 32B, 2 positions per lane,
    // each position = two B128 transfers (IOFFSET applies to both sides).
    #pragma unroll
    for (int q = 0; q < 2; ++q) {
      int pos = lane + q * 32;
      int py = pos >> 3, px = pos & 7;
      const half_t* src = in + (((size_t)b * H_ + (y0 - 2 + py)) * W_ + (x0 - 2 + px)) * 16;
      unsigned long long ga = (unsigned long long)(size_t)src;
      unsigned ldst = (unsigned)(size_t)(L + (size_t)pos * 16);
      asm volatile("global_load_async_to_lds_b128 %0, %1, off\n\t"
                   "global_load_async_to_lds_b128 %0, %1, off offset:16"
                   :: "v"(ldst), "v"(ga) : "memory");
    }
    asm volatile("s_wait_asynccnt 0x0" ::: "memory");
  } else {
    // border: predicated VGPR round-trip with zero fill
    for (int q = 0; q < 2; ++q) {
      int pos = lane + q * 32;
      int py = pos >> 3, px = pos & 7;
      int gy = y0 - 2 + py, gx = x0 - 2 + px;
      int4 v0 = {0, 0, 0, 0}, v1 = {0, 0, 0, 0};
      if (gy >= 0 && gy < H_ && gx >= 0 && gx < W_) {
        const int4* src = (const int4*)(in + (((size_t)b * H_ + gy) * W_ + gx) * 16);
        v0 = src[0]; v1 = src[1];
      }
      int4* d = (int4*)(L + (size_t)pos * 16);
      d[0] = v0; d[1] = v1;
    }
    // DS ops are in-order within a wave on CDNA5; compiler inserts dscnt waits.
  }

  const int m   = lane & 15;
  const int my  = m >> 2, mx = m & 3;
  const int cho = (lane < 16) ? 0 : 8;  // A channel sub-block per lane half
  const int n   = lane & 15;            // output channel (B column)
  v8f acc = {0.f, 0.f, 0.f, 0.f, 0.f, 0.f, 0.f, 0.f};

  for (int s = 0; s < 13; ++s) {
    int t0 = 2 * s, t1 = 2 * s + 1;
    int t1c = (t1 > 24) ? 24 : t1;  // padded tap: weight is zero anyway
    int dy0 = t0 / 5, dx0 = t0 % 5, dy1 = t1c / 5, dx1 = t1c % 5;
    v16h a, bf;
    ((int4*)&a)[0] = *(const int4*)(L + ((my + dy0) * 8 + (mx + dx0)) * 16 + cho);
    ((int4*)&a)[1] = *(const int4*)(L + ((my + dy1) * 8 + (mx + dx1)) * 16 + cho);
    // B rows: elems 0-7 tap t0 ch cho..cho+7, elems 8-15 tap t1 ch cho..cho+7 (matches A)
    ((int4*)&bf)[0] = *(const int4*)(wp + ((size_t)t0 * 16 + n) * 16 + cho);
    ((int4*)&bf)[1] = *(const int4*)(wp + ((size_t)t1 * 16 + n) * 16 + cho);
    acc = wmma_f16(a, bf, acc);
  }
  float bs = bias[n];
  #pragma unroll
  for (int v = 0; v < 8; ++v) {
    int mm = v + ((lane < 16) ? 0 : 8);
    int oy = y0 + (mm >> 2), ox = x0 + (mm & 3);
    out[(((size_t)b * 16 + n) * H_ + oy) * W_ + ox] = acc[v] + bs;
  }
}

// ------------------------------- orientation -------------------------------
__global__ void k_orient(const half_t* __restrict__ feat, const float* __restrict__ w,
                         const float* __restrict__ bias, float* __restrict__ ori) {
  int i = blockIdx.x * blockDim.x + threadIdx.x;
  if (i >= B_ * HW_) return;
  int b = i / HW_, p = i % HW_, y = p / W_, x = p % W_;
  float a0 = bias[0], a1 = bias[1];
  for (int ky = 0; ky < 5; ++ky) {
    int yy = y + ky - 2; if (yy < 0 || yy >= H_) continue;
    for (int kx = 0; kx < 5; ++kx) {
      int xx = x + kx - 2; if (xx < 0 || xx >= W_) continue;
      const half_t* s = feat + (((size_t)b * H_ + yy) * W_ + xx) * 16;
      int t = ky * 5 + kx;
      for (int c = 0; c < 16; ++c) {
        float v = (float)s[c];
        a0 += w[c * 25 + t] * v;
        a1 += w[(16 + c) * 25 + t] * v;
      }
    }
  }
  float nrm = sqrtf(a0 * a0 + a1 * a1);
  float r = (nrm > 0.f) ? (1.f / nrm) : 0.f;
  ori[2 * (size_t)i] = a0 * r; ori[2 * (size_t)i + 1] = a1 * r;
}

// ------------------------------ scale pyramid ------------------------------
__global__ void k_resize_feat(const half_t* __restrict__ feat, half_t* __restrict__ r,
                              int oh, int ow) {
  int i = blockIdx.x * blockDim.x + threadIdx.x;
  if (i >= B_ * oh * ow) return;
  int b = i / (oh * ow), p = i % (oh * ow), oy = p / ow, ox = p % ow;
  float sy = fmaxf((oy + 0.5f) * ((float)H_ / oh) - 0.5f, 0.f);
  float sx = fmaxf((ox + 0.5f) * ((float)W_ / ow) - 0.5f, 0.f);
  int y0 = min((int)sy, H_ - 1), y1 = min(y0 + 1, H_ - 1);
  int x0 = min((int)sx, W_ - 1), x1 = min(x0 + 1, W_ - 1);
  float wy = sy - (float)y0, wx = sx - (float)x0;
  const half_t* p00 = feat + (((size_t)b * H_ + y0) * W_ + x0) * 16;
  const half_t* p01 = feat + (((size_t)b * H_ + y0) * W_ + x1) * 16;
  const half_t* p10 = feat + (((size_t)b * H_ + y1) * W_ + x0) * 16;
  const half_t* p11 = feat + (((size_t)b * H_ + y1) * W_ + x1) * 16;
  half_t* d = r + (size_t)i * 16;
  for (int c = 0; c < 16; ++c) {
    float top = (1.f - wx) * (float)p00[c] + wx * (float)p01[c];
    float bot = (1.f - wx) * (float)p10[c] + wx * (float)p11[c];
    d[c] = (half_t)((1.f - wy) * top + wy * bot);
  }
}

__global__ void k_conv_scale(const half_t* __restrict__ r, const float* __restrict__ w,
                             const float* __restrict__ bias, float* __restrict__ sm,
                             int oh, int ow) {
  int i = blockIdx.x * blockDim.x + threadIdx.x;
  if (i >= B_ * oh * ow) return;
  int b = i / (oh * ow), p = i % (oh * ow), y = p / ow, x = p % ow;
  float acc = bias[0];
  for (int ky = 0; ky < 5; ++ky) {
    int yy = y + ky - 2; if (yy < 0 || yy >= oh) continue;
    for (int kx = 0; kx < 5; ++kx) {
      int xx = x + kx - 2; if (xx < 0 || xx >= ow) continue;
      const half_t* s = r + (((size_t)b * oh + yy) * ow + xx) * 16;
      int t = ky * 5 + kx;
      for (int c = 0; c < 16; ++c) acc += w[c * 25 + t] * (float)s[c];
    }
  }
  sm[i] = acc;
}

__global__ void k_stats(const float* __restrict__ sm, float* __restrict__ stats, int n) {
  __shared__ float s1[256], s2[256];
  int b = blockIdx.x;
  float a = 0.f, q = 0.f;
  for (int j = threadIdx.x; j < n; j += 256) {
    float v = sm[(size_t)b * n + j];
    a += v; q += v * v;
  }
  s1[threadIdx.x] = a; s2[threadIdx.x] = q;
  __syncthreads();
  for (int st = 128; st > 0; st >>= 1) {
    if ((int)threadIdx.x < st) { s1[threadIdx.x] += s1[threadIdx.x + st]; s2[threadIdx.x] += s2[threadIdx.x + st]; }
    __syncthreads();
  }
  if (threadIdx.x == 0) {
    float m = s1[0] / n, var = s2[0] / n - m * m;
    stats[b * 2] = m; stats[b * 2 + 1] = rsqrtf(var + 1e-3f);
  }
}

// instance-norm fused into upsample (linear ops commute)
__global__ void k_upmap(const float* __restrict__ sm, const float* __restrict__ stats,
                        float* __restrict__ maps, int oh, int ow, int si) {
  int i = blockIdx.x * blockDim.x + threadIdx.x;
  if (i >= B_ * HW_) return;
  int b = i / HW_, p = i % HW_, y = p / W_, x = p % W_;
  float sy = fmaxf((y + 0.5f) * ((float)oh / H_) - 0.5f, 0.f);
  float sx = fmaxf((x + 0.5f) * ((float)ow / W_) - 0.5f, 0.f);
  int y0 = min((int)sy, oh - 1), y1 = min(y0 + 1, oh - 1);
  int x0 = min((int)sx, ow - 1), x1 = min(x0 + 1, ow - 1);
  float wy = sy - y0, wx = sx - x0;
  const float* s = sm + (size_t)b * oh * ow;
  float top = (1.f - wx) * s[y0 * ow + x0] + wx * s[y0 * ow + x1];
  float bot = (1.f - wx) * s[y1 * ow + x0] + wx * s[y1 * ow + x1];
  float v = (1.f - wy) * top + wy * bot;
  maps[((size_t)b * NS + si) * HW_ + p] = (v - stats[b * 2]) * stats[b * 2 + 1];
}

// --------------------------- softmaxpool / NMS -----------------------------
__global__ void k_chmax(const float* __restrict__ maps, float* __restrict__ o) {
  int i = blockIdx.x * blockDim.x + threadIdx.x;
  if (i >= B_ * HW_) return;
  int b = i / HW_, p = i % HW_;
  float m = -1e30f;
  for (int s = 0; s < NS; ++s) m = fmaxf(m, maps[((size_t)b * NS + s) * HW_ + p]);
  o[i] = m;
}
__global__ void k_hmax(const float* __restrict__ in, float* __restrict__ o) {
  int i = blockIdx.x * blockDim.x + threadIdx.x;
  if (i >= B_ * HW_) return;
  int b = i / HW_, p = i % HW_, y = p / W_, x = p % W_;
  float m = -1e30f;
  for (int dx = -7; dx <= 7; ++dx) { int xx = x + dx; if (xx >= 0 && xx < W_) m = fmaxf(m, in[(size_t)b * HW_ + y * W_ + xx]); }
  o[i] = m;
}
__global__ void k_vmax(const float* __restrict__ in, float* __restrict__ o) {
  int i = blockIdx.x * blockDim.x + threadIdx.x;
  if (i >= B_ * HW_) return;
  int b = i / HW_, p = i % HW_, y = p / W_, x = p % W_;
  float m = -1e30f;
  for (int dy = -7; dy <= 7; ++dy) { int yy = y + dy; if (yy >= 0 && yy < H_) m = fmaxf(m, in[(size_t)b * HW_ + yy * W_ + x]); }
  o[i] = m;
}
__global__ void k_es(const float* __restrict__ maps, const float* __restrict__ nmp,
                     float* __restrict__ o) {
  int i = blockIdx.x * blockDim.x + threadIdx.x;
  if (i >= B_ * HW_) return;
  int b = i / HW_, p = i % HW_;
  float nm = nmp[i], acc = 0.f;
  for (int s = 0; s < NS; ++s) acc += expf(3.f * (maps[((size_t)b * NS + s) * HW_ + p] - nm));
  o[i] = acc;
}
__global__ void k_boxh(const float* __restrict__ in, float* __restrict__ o) {
  int i = blockIdx.x * blockDim.x + threadIdx.x;
  if (i >= B_ * HW_) return;
  int b = i / HW_, p = i % HW_, y = p / W_, x = p % W_;
  float a = 0.f;
  for (int dx = -7; dx <= 7; ++dx) { int xx = x + dx; if (xx >= 0 && xx < W_) a += in[(size_t)b * HW_ + y * W_ + xx]; }
  o[i] = a;
}
__global__ void k_boxv(const float* __restrict__ in, float* __restrict__ o) {
  int i = blockIdx.x * blockDim.x + threadIdx.x;
  if (i >= B_ * HW_) return;
  int b = i / HW_, p = i % HW_, y = p / W_, x = p % W_;
  float a = 0.f;
  for (int dy = -7; dy <= 7; ++dy) { int yy = y + dy; if (yy >= 0 && yy < H_) a += in[(size_t)b * HW_ + yy * W_ + x]; }
  o[i] = a;
}
__global__ void k_soft(const float* __restrict__ maps, const float* __restrict__ nmp,
                       const float* __restrict__ sums, float* __restrict__ mh,
                       float* __restrict__ msc) {
  int i = blockIdx.x * blockDim.x + threadIdx.x;
  if (i >= B_ * HW_) return;
  int b = i / HW_, p = i % HW_, y = p / W_, x = p % W_;
  float nm = nmp[i], den = sums[i] + 1e-6f;
  float h[NS], hmax = -1e30f;
  for (int s = 0; s < NS; ++s) {
    h[s] = expf(3.f * (maps[((size_t)b * NS + s) * HW_ + p] - nm)) / den;
    hmax = fmaxf(hmax, h[s]);
  }
  float esum = 0.f, hacc = 0.f, sacc = 0.f;
  for (int s = 0; s < NS; ++s) {
    float e = expf(100.f * (h[s] - hmax));
    esum += e; hacc += h[s] * e; sacc += c_sf[s] * e;
  }
  float inv = 1.f / (esum + 1e-8f);
  bool in10 = (y >= 10 && y < H_ - 10 && x >= 10 && x < W_ - 10);
  mh[i] = in10 ? hacc * inv : 0.f;
  msc[i] = sacc * inv;
}
__global__ void k_nms(const float* __restrict__ mh, float* __restrict__ sc) {
  int i = blockIdx.x * blockDim.x + threadIdx.x;
  if (i >= B_ * HW_) return;
  int b = i / HW_, p = i % HW_, y = p / W_, x = p % W_;
  float c = mh[i];
  bool peak = true;
  for (int dy = -2; dy <= 2; ++dy)
    for (int dx = -2; dx <= 2; ++dx) {
      if (dy == 0 && dx == 0) continue;
      int yy = y + dy, xx = x + dx;
      float v = (yy >= 0 && yy < H_ && xx >= 0 && xx < W_) ? mh[(size_t)b * HW_ + yy * W_ + xx] : 0.f;
      peak = peak && (c > v);
    }
  bool in16 = (y >= 16 && y < H_ - 16 && x >= 16 && x < W_ - 16);
  sc[i] = (peak && in16) ? c : 0.f;
}

// ---------------- deterministic per-image top-k (one wave / image) ---------
__global__ void __launch_bounds__(32) k_select(const float* __restrict__ scores,
                                               int* __restrict__ kidx) {
  __shared__ int hist[1024];
  __shared__ int s_thr, cnt;
  int b = blockIdx.x, lane = threadIdx.x;
  const float* s = scores + (size_t)b * HW_;
  for (int j = lane; j < 1024; j += 32) hist[j] = 0;
  for (int j = lane; j < TOPK; j += 32) kidx[b * TOPK + j] = 0;  // pad default
  __syncthreads();
  for (int j = lane; j < HW_; j += 32) {
    float v = s[j];
    if (v > 0.f) {
      int bin = min((int)(__float_as_uint(v) >> 20), 1023);
      atomicAdd(&hist[bin], 1);
    }
  }
  __syncthreads();
  if (lane == 0) {
    int acc = 0, thr = -1;
    for (int bin = 1023; bin >= 0; --bin) {
      int h = hist[bin];
      if (acc + h >= TOPK) { thr = bin; break; }
      acc += h;
    }
    s_thr = thr; cnt = 0;
  }
  __syncthreads();
  int thr = s_thr;
  for (int phase = 0; phase < 2; ++phase) {  // phase0: bin>thr, phase1: bin==thr
    for (int base = 0; base < HW_; base += 32) {
      int j = base + lane;
      bool pred = false;
      float v = s[j];
      if (v > 0.f) {
        int bin = min((int)(__float_as_uint(v) >> 20), 1023);
        pred = phase == 0 ? (bin > thr) : (bin == thr);
      }
      unsigned long long bal = __ballot(pred);
      int pre = __popcll(bal & ((1ull << lane) - 1ull));
      int tot = __popcll(bal);
      int base_cnt = cnt;  // read before lane-0 update (wave in-order)
      if (pred) {
        int pos = base_cnt + pre;
        if (pos < TOPK) kidx[b * TOPK + pos] = j;
      }
      if (lane == 0) cnt = base_cnt + tot;
      __syncthreads();
    }
  }
}

__global__ void k_gather(const int* __restrict__ kidx, const float* __restrict__ msc,
                         const float* __restrict__ ori, float* __restrict__ dout,
                         float* __restrict__ kscale, float* __restrict__ kori) {
  int n = blockIdx.x * blockDim.x + threadIdx.x;
  if (n >= NKP) return;
  int b = n / TOPK, idx = kidx[n];
  int kx = idx % W_, ky = idx / W_;
  dout[2 * n] = (float)ky; dout[2 * n + 1] = (float)kx;  // yx_kpts
  kscale[n] = msc[(size_t)b * HW_ + idx];
  kori[2 * n]     = ori[((size_t)b * HW_ + idx) * 2];
  kori[2 * n + 1] = ori[((size_t)b * HW_ + idx) * 2 + 1];
}

// ------------------------------- patch crop --------------------------------
__global__ void k_crop(const float* __restrict__ data, const int* __restrict__ kidx,
                       const float* __restrict__ kscale, const float* __restrict__ kori,
                       half_t* __restrict__ patches) {
  int i = blockIdx.x * blockDim.x + threadIdx.x;
  if (i >= NKP * 1024) return;
  int n = i >> 10, p = i & 1023, oy = p >> 5, ox = p & 31;
  int b = n / TOPK, idx = kidx[n];
  float kx = (float)(idx % W_), ky = (float)(idx / W_);
  float gx = -1.f + 2.f * ox / 31.f, gy = -1.f + 2.f * oy / 31.f;
  float cs = kori[2 * n], sn = kori[2 * n + 1], sc = kscale[n];
  float X = sc * (cs * gx - sn * gy) * 16.f + kx;
  float Y = sc * (sn * gx + cs * gy) * 16.f + ky;
  float x0f = floorf(X), y0f = floorf(Y);
  int x0 = min(max((int)x0f, 0), W_ - 1), x1 = min(max((int)x0f + 1, 0), W_ - 1);
  int y0 = min(max((int)y0f, 0), H_ - 1), y1 = min(max((int)y0f + 1, 0), H_ - 1);
  const float* db = data + (size_t)b * HW_;
  float Ia = db[y0 * W_ + x0], Ib = db[y1 * W_ + x0];
  float Ic = db[y0 * W_ + x1], Id = db[y1 * W_ + x1];
  float wa = (x0f + 1.f - X) * (y0f + 1.f - Y), wb = (x0f + 1.f - X) * (Y - y0f);
  float wc = (X - x0f) * (y0f + 1.f - Y),       wd = (X - x0f) * (Y - y0f);
  patches[i] = (half_t)(wa * Ia + wb * Ib + wc * Ic + wd * Id);
}

// --------------------------- descriptor network ----------------------------
// conv1 1->64 s2 (scalar; tiny), with bn+leaky fused -> NHWC f16 [n][16][16][64]
__global__ void k_dconv1(const half_t* __restrict__ patches, const float* __restrict__ w,
                         const float* __restrict__ bias, const float* g, const float* bb,
                         const float* mm, const float* vv, half_t* __restrict__ act1) {
  int i = blockIdx.x * blockDim.x + threadIdx.x;
  if (i >= NKP * 256) return;
  int n = i >> 8, p = i & 255, oy = p >> 4, ox = p & 15;
  float vin[9];
  for (int ky = 0; ky < 3; ++ky)
    for (int kx = 0; kx < 3; ++kx) {
      int yy = 2 * oy - 1 + ky, xx = 2 * ox - 1 + kx;
      vin[ky * 3 + kx] = (yy >= 0 && yy < 32 && xx >= 0 && xx < 32)
                             ? (float)patches[((size_t)n * 32 + yy) * 32 + xx] : 0.f;
    }
  half_t* d = act1 + (size_t)i * 64;
  for (int o = 0; o < 64; ++o) {
    float a = bias[o];
    for (int t = 0; t < 9; ++t) a += w[o * 9 + t] * vin[t];
    d[o] = (half_t)leakyf(bnf(a, g[o], bb[o], mm[o], vv[o]));
  }
}

// im2col (stride2, pad1, 3x3), K order = i*9 + tap  (matches OIHW weight flat)
__global__ void k_im2col(const half_t* __restrict__ src, half_t* __restrict__ A,
                         int hs, int os, int ci) {
  int m = blockIdx.x * blockDim.x + threadIdx.x;
  if (m >= NKP * os * os) return;
  int n = m / (os * os), p = m % (os * os), y = p / os, x = p % os;
  half_t* a = A + (size_t)m * (ci * 9);
  for (int ky = 0; ky < 3; ++ky)
    for (int kx = 0; kx < 3; ++kx) {
      int yy = 2 * y - 1 + ky, xx = 2 * x - 1 + kx, t = ky * 3 + kx;
      if (yy >= 0 && yy < hs && xx >= 0 && xx < hs) {
        const half_t* s = src + (((size_t)n * hs + yy) * hs + xx) * ci;
        for (int c = 0; c < ci; ++c) a[c * 9 + t] = s[c];
      } else {
        for (int c = 0; c < ci; ++c) a[c * 9 + t] = (half_t)0.f;
      }
    }
}

// generic WMMA GEMM: C[M][N] = A[M][K] * Bw[N][K]^T, f16 in, f32 out
__global__ void __launch_bounds__(256) k_gemm(const half_t* __restrict__ A,
                                              const half_t* __restrict__ Bw,
                                              float* __restrict__ C, int M, int N, int K) {
  int wave = threadIdx.x >> 5, lane = threadIdx.x & 31;
  long tile = (long)blockIdx.x * 8 + wave;
  int ntn = N >> 4;
  long ntiles = (long)(M >> 4) * ntn;
  if (tile >= ntiles) return;
  int mt = (int)(tile / ntn), nt = (int)(tile % ntn);
  int r  = lane & 15;
  int hi = lane >> 4;  // 0 or 1
  const half_t* ap = A  + (size_t)(mt * 16 + r) * K + hi * 8;
  const half_t* bp = Bw + (size_t)(nt * 16 + r) * K + hi * 8;
  v8f acc = {0.f, 0.f, 0.f, 0.f, 0.f, 0.f, 0.f, 0.f};
  for (int k0 = 0; k0 < K; k0 += 32) {
    __builtin_prefetch(ap + k0 + 256, 0, 0);
    v16h a, b;
    ((int4*)&a)[0] = *(const int4*)(ap + k0);
    ((int4*)&a)[1] = *(const int4*)(ap + k0 + 16);
    ((int4*)&b)[0] = *(const int4*)(bp + k0);
    ((int4*)&b)[1] = *(const int4*)(bp + k0 + 16);
    acc = wmma_f16(a, b, acc);
  }
  float* cp = C + (size_t)(mt * 16 + hi * 8) * N + nt * 16 + r;
  #pragma unroll
  for (int v = 0; v < 8; ++v) cp[(size_t)v * N] = acc[v];
}

// bias (+optional bn) (+optional leaky), writes f16 and/or f32
__global__ void k_post(const float* __restrict__ raw, const float* __restrict__ bias,
                       const float* g, const float* bb, const float* mm, const float* vv,
                       half_t* dst16, float* dst32, long total, int N, int act) {
  long i = (long)blockIdx.x * blockDim.x + threadIdx.x;
  if (i >= total) return;
  int n = (int)(i % N);
  float x = raw[i] + bias[n];
  if (g) x = bnf(x, g[n], bb[n], mm[n], vv[n]);
  if (act) x = leakyf(x);
  if (dst16) dst16[i] = (half_t)x;
  if (dst32) dst32[i] = x;
}

// NHWC [n][4][4][256] -> flat [n][k], k = c*16 + y*4 + x (reference reshape)
__global__ void k_flat(const half_t* __restrict__ act3, half_t* __restrict__ A) {
  int i = blockIdx.x * blockDim.x + threadIdx.x;
  if (i >= NKP * 4096) return;
  int n = i >> 12, k = i & 4095, c = k >> 4, yx = k & 15, y = yx >> 2, x = yx & 3;
  A[i] = act3[(((size_t)n * 4 + y) * 4 + x) * 256 + c];
}

__global__ void k_l2(const float* __restrict__ v, float* __restrict__ dout) {
  __shared__ float red[256];
  int n = blockIdx.x, t = threadIdx.x;
  float x = v[(size_t)n * 256 + t];
  red[t] = x * x;
  __syncthreads();
  for (int st = 128; st > 0; st >>= 1) {
    if (t < st) red[t] += red[t + st];
    __syncthreads();
  }
  float r = rsqrtf(red[0]);
  dout[2 * NKP + (size_t)n * 256 + t] = x * r;
}

// ---------------------------------------------------------------------------
// host side
// ---------------------------------------------------------------------------

struct BNp { const float *g, *b, *m, *v; };
struct Params {
  const float* x;
  const float *init_w, *init_b;
  struct Blk { BNp pre, mid; const float *w1, *b1, *w2, *b2; } blk[3];
  BNp fin;
  const float *sw[5], *sb[5];
  const float *ori_w, *ori_b;
  struct Dsc { const float *w, *b; BNp bn; } dsc[3];
  const float *fc1w, *fc1b, *fc2w, *fc2b;
  BNp fcbn;
};

extern "C" void kernel_launch(void* const* d_in, const int* in_sizes, int n_in,
                              void* d_out, int out_size, void* d_ws, size_t ws_size,
                              hipStream_t stream) {
  const float* const* F = (const float* const*)d_in;
  Params P;
  if (in_sizes[0] == B_ * 3 * HW_) {
    // insertion-order flatten: x first, then params in dict insertion order
    int k = 0;
    P.x = F[k++];
    P.init_w = F[k++]; P.init_b = F[k++];
    for (int b = 0; b < 3; ++b) {
      P.blk[b].pre = {F[k], F[k + 1], F[k + 2], F[k + 3]}; k += 4;  // g,b,m,v
      P.blk[b].w1 = F[k++]; P.blk[b].b1 = F[k++];
      P.blk[b].mid = {F[k], F[k + 1], F[k + 2], F[k + 3]}; k += 4;
      P.blk[b].w2 = F[k++]; P.blk[b].b2 = F[k++];
    }
    P.fin = {F[k], F[k + 1], F[k + 2], F[k + 3]}; k += 4;
    for (int s = 0; s < 5; ++s) { P.sw[s] = F[k++]; P.sb[s] = F[k++]; }
    P.ori_w = F[k++]; P.ori_b = F[k++];
    for (int d = 0; d < 3; ++d) {
      P.dsc[d].w = F[k++]; P.dsc[d].b = F[k++];
      P.dsc[d].bn = {F[k], F[k + 1], F[k + 2], F[k + 3]}; k += 4;
    }
    P.fc1w = F[k++]; P.fc1b = F[k++];
    P.fcbn = {F[k + 1], F[k], F[k + 2], F[k + 3]}; k += 4;  // dict insertion b,g,m,v
    P.fc2w = F[k++]; P.fc2b = F[k++];
  } else {
    // jax pytree (sorted-key) flatten: params leaves first (alphabetical), then x
    int k = 0;
    for (int b = 0; b < 3; ++b) {  // bb_i: conv1{b,w} conv2{b,w} mid_bn{b,g,m,v} pre_bn{b,g,m,v}
      P.blk[b].b1 = F[k]; P.blk[b].w1 = F[k + 1];
      P.blk[b].b2 = F[k + 2]; P.blk[b].w2 = F[k + 3];
      P.blk[b].mid = {F[k + 5], F[k + 4], F[k + 6], F[k + 7]};
      P.blk[b].pre = {F[k + 9], F[k + 8], F[k + 10], F[k + 11]};
      k += 12;
    }
    for (int d = 0; d < 3; ++d) {  // desc[d]: b, bn{b,g,m,v}, w
      P.dsc[d].b = F[k];
      P.dsc[d].bn = {F[k + 2], F[k + 1], F[k + 3], F[k + 4]};
      P.dsc[d].w = F[k + 5];
      k += 6;
    }
    P.fc1b = F[k]; P.fc1w = F[k + 1]; k += 2;
    P.fc2b = F[k]; P.fc2w = F[k + 1]; k += 2;
    P.fcbn = {F[k + 1], F[k], F[k + 2], F[k + 3]}; k += 4;   // fc_bn1 b,g,m,v
    P.fin  = {F[k + 1], F[k], F[k + 2], F[k + 3]}; k += 4;   // fin_bn
    P.init_b = F[k]; P.init_w = F[k + 1]; k += 2;
    P.ori_b = F[k]; P.ori_w = F[k + 1]; k += 2;
    for (int s = 0; s < 5; ++s) { P.sb[s] = F[k]; P.sw[s] = F[k + 1]; k += 2; }
    P.x = F[k];
  }

  // workspace bump allocator (assumes generously sized scratch; compile-only env)
  char* ws = (char*)d_ws;
  size_t off = 0;
  auto AL = [&](size_t bytes) -> void* {
    void* p = ws + off;
    off = (off + bytes + 255) & ~(size_t)255;
    return p;
  };
  float*  data   = (float*)AL(sizeof(float) * B_ * HW_);
  float*  f      = (float*)AL(sizeof(float) * B_ * 16 * HW_);
  float*  cvt    = (float*)AL(sizeof(float) * B_ * 16 * HW_);
  half_t* h16    = (half_t*)AL(sizeof(half_t) * B_ * 16 * HW_);
  half_t* feat16 = (half_t*)AL(sizeof(half_t) * B_ * 16 * HW_);
  float*  ori    = (float*)AL(sizeof(float) * B_ * HW_ * 2);
  half_t* wp[6];
  for (int i = 0; i < 6; ++i) wp[i] = (half_t*)AL(sizeof(half_t) * 26 * 256);
  half_t* r16   = (half_t*)AL(sizeof(half_t) * (size_t)B_ * 678 * 905 * 16);
  float*  smraw = (float*)AL(sizeof(float) * (size_t)B_ * 678 * 905);
  float*  stats = (float*)AL(sizeof(float) * 2 * B_);
  float*  maps  = (float*)AL(sizeof(float) * (size_t)B_ * NS * HW_);
  float*  tA    = (float*)AL(sizeof(float) * B_ * HW_);
  float*  tB    = (float*)AL(sizeof(float) * B_ * HW_);
  float*  nmp   = (float*)AL(sizeof(float) * B_ * HW_);
  float*  sums  = (float*)AL(sizeof(float) * B_ * HW_);
  float*  mh    = (float*)AL(sizeof(float) * B_ * HW_);
  float*  msc   = (float*)AL(sizeof(float) * B_ * HW_);
  float*  scr   = (float*)AL(sizeof(float) * B_ * HW_);
  int*    kidx  = (int*)AL(sizeof(int) * NKP);
  float*  kscale = (float*)AL(sizeof(float) * NKP);
  float*  kori  = (float*)AL(sizeof(float) * 2 * NKP);
  half_t* patches = (half_t*)AL(sizeof(half_t) * (size_t)NKP * 1024);
  half_t* act1 = (half_t*)AL(sizeof(half_t) * (size_t)NKP * 256 * 64);
  half_t* A2   = (half_t*)AL(sizeof(half_t) * 128000ull * 576);
  float*  raw2 = (float*)AL(sizeof(float) * 128000ull * 128);
  half_t* act2 = (half_t*)AL(sizeof(half_t) * 128000ull * 128);
  half_t* A3   = (half_t*)AL(sizeof(half_t) * 32000ull * 1152);
  float*  raw3 = (float*)AL(sizeof(float) * 32000ull * 256);
  half_t* act3 = (half_t*)AL(sizeof(half_t) * 32000ull * 256);
  half_t* Afc  = (half_t*)AL(sizeof(half_t) * 2000ull * 4096);
  float*  rawf1 = (float*)AL(sizeof(float) * 2000ull * 512);
  half_t* actf1 = (half_t*)AL(sizeof(half_t) * 2000ull * 512);
  float*  rawf2 = (float*)AL(sizeof(float) * 2000ull * 256);
  half_t* wd2 = (half_t*)AL(sizeof(half_t) * 128 * 576);
  half_t* wd3 = (half_t*)AL(sizeof(half_t) * 256 * 1152);
  half_t* wf1 = (half_t*)AL(sizeof(half_t) * 512 * 4096);
  half_t* wf2 = (half_t*)AL(sizeof(half_t) * 256 * 512);
  (void)ws_size; (void)n_in; (void)out_size;

  float* dout = (float*)d_out;
  const int T = 256;
  auto G = [](long n) { return (int)((n + 255) / 256); };

  // ---- backbone ----
  k_mean3<<<G(B_ * HW_), T, 0, stream>>>(P.x, data);
  k_conv_init<<<G(B_ * HW_), T, 0, stream>>>(data, P.init_w, P.init_b, f);
  const float* bw[6] = {P.blk[0].w1, P.blk[0].w2, P.blk[1].w1, P.blk[1].w2, P.blk[2].w1, P.blk[2].w2};
  for (int i = 0; i < 6; ++i) k_pack_w5<<<G(26 * 256), T, 0, stream>>>(bw[i], wp[i]);

  long nf = (long)B_ * 16 * HW_;
  for (int b = 0; b < 3; ++b) {
    k_bn_act_nhwc<<<G(nf), T, 0, stream>>>(f, P.blk[b].pre.g, P.blk[b].pre.b, P.blk[b].pre.m, P.blk[b].pre.v, h16);
    k_conv5x5_wmma<<<9600, T, 0, stream>>>(h16, wp[2 * b], P.blk[b].b1, cvt);
    k_bn_act_nhwc<<<G(nf), T, 0, stream>>>(cvt, P.blk[b].mid.g, P.blk[b].mid.b, P.blk[b].mid.m, P.blk[b].mid.v, h16);
    k_conv5x5_wmma<<<9600, T, 0, stream>>>(h16, wp[2 * b + 1], P.blk[b].b2, cvt);
    k_add<<<G(nf), T, 0, stream>>>(f, cvt, nf);
  }
  k_bn_act_nhwc<<<G(nf), T, 0, stream>>>(f, P.fin.g, P.fin.b, P.fin.m, P.fin.v, feat16);
  k_orient<<<G(B_ * HW_), T, 0, stream>>>(feat16, P.ori_w, P.ori_b, ori);

  // ---- scale pyramid ----
  const int OHs[5] = {339, 403, 480, 570, 678};
  const int OWs[5] = {452, 538, 640, 761, 905};
  for (int s = 0; s < 5; ++s) {
    int oh = OHs[s], ow = OWs[s];
    long n = (long)B_ * oh * ow;
    k_resize_feat<<<G(n), T, 0, stream>>>(feat16, r16, oh, ow);
    k_conv_scale<<<G(n), T, 0, stream>>>(r16, P.sw[s], P.sb[s], smraw, oh, ow);
    k_stats<<<B_, T, 0, stream>>>(smraw, stats, oh * ow);
    k_upmap<<<G(B_ * HW_), T, 0, stream>>>(smraw, stats, maps, oh, ow, s);
  }

  // ---- softmaxpool / soft-argmax / NMS / top-k ----
  int gb = G(B_ * HW_);
  k_chmax<<<gb, T, 0, stream>>>(maps, tA);
  k_hmax<<<gb, T, 0, stream>>>(tA, tB);
  k_vmax<<<gb, T, 0, stream>>>(tB, nmp);
  k_es<<<gb, T, 0, stream>>>(maps, nmp, tA);
  k_boxh<<<gb, T, 0, stream>>>(tA, tB);
  k_boxv<<<gb, T, 0, stream>>>(tB, sums);
  k_soft<<<gb, T, 0, stream>>>(maps, nmp, sums, mh, msc);
  k_nms<<<gb, T, 0, stream>>>(mh, scr);
  k_select<<<B_, 32, 0, stream>>>(scr, kidx);
  k_gather<<<G(NKP), T, 0, stream>>>(kidx, msc, ori, dout, kscale, kori);

  // ---- crop + descriptor ----
  k_crop<<<G((long)NKP * 1024), T, 0, stream>>>(data, kidx, kscale, kori, patches);
  k_dconv1<<<G((long)NKP * 256), T, 0, stream>>>(patches, P.dsc[0].w, P.dsc[0].b,
      P.dsc[0].bn.g, P.dsc[0].bn.b, P.dsc[0].bn.m, P.dsc[0].bn.v, act1);

  k_tof16<<<G(128 * 576), T, 0, stream>>>(P.dsc[1].w, wd2, 128 * 576);
  k_tof16<<<G(256 * 1152), T, 0, stream>>>(P.dsc[2].w, wd3, 256 * 1152);
  k_tof16<<<G(512 * 4096), T, 0, stream>>>(P.fc1w, wf1, 512 * 4096);
  k_tof16<<<G(256 * 512), T, 0, stream>>>(P.fc2w, wf2, 256 * 512);

  // conv2: M=128000, K=576, N=128
  k_im2col<<<G(128000), T, 0, stream>>>(act1, A2, 16, 8, 64);
  k_gemm<<<(128000 / 16) * (128 / 16) / 8, T, 0, stream>>>(A2, wd2, raw2, 128000, 128, 576);
  k_post<<<G(128000ll * 128), T, 0, stream>>>(raw2, P.dsc[1].b, P.dsc[1].bn.g, P.dsc[1].bn.b,
      P.dsc[1].bn.m, P.dsc[1].bn.v, act2, nullptr, 128000ll * 128, 128, 1);

  // conv3: M=32000, K=1152, N=256
  k_im2col<<<G(32000), T, 0, stream>>>(act2, A3, 8, 4, 128);
  k_gemm<<<(32000 / 16) * (256 / 16) / 8, T, 0, stream>>>(A3, wd3, raw3, 32000, 256, 1152);
  k_post<<<G(32000ll * 256), T, 0, stream>>>(raw3, P.dsc[2].b, P.dsc[2].bn.g, P.dsc[2].bn.b,
      P.dsc[2].bn.m, P.dsc[2].bn.v, act3, nullptr, 32000ll * 256, 256, 1);

  // fc1: M=2000, K=4096, N=512
  k_flat<<<G((long)NKP * 4096), T, 0, stream>>>(act3, Afc);
  k_gemm<<<(2000 / 16) * (512 / 16) / 8, T, 0, stream>>>(Afc, wf1, rawf1, 2000, 512, 4096);
  k_post<<<G(2000ll * 512), T, 0, stream>>>(rawf1, P.fc1b, P.fcbn.g, P.fcbn.b,
      P.fcbn.m, P.fcbn.v, actf1, nullptr, 2000ll * 512, 512, 1);

  // fc2: M=2000, K=512, N=256
  k_gemm<<<(2000 / 16) * (256 / 16) / 8, T, 0, stream>>>(actf1, wf2, rawf2, 2000, 256, 512);
  k_post<<<G(2000ll * 256), T, 0, stream>>>(rawf2, P.fc2b, nullptr, nullptr, nullptr, nullptr,
      nullptr, rawf2, 2000ll * 256, 256, 0);
  k_l2<<<NKP, 256, 0, stream>>>(rawf2, dout);
}
